// mixture_of_GCNs_1056561954825
// MI455X (gfx1250) — matrix operand order repaired and checked by reading, not compiled, not run on hardware
//
#include <hip/hip_runtime.h>

// ---------------------------------------------------------------------------
// Problem constants (match the reference)
// ---------------------------------------------------------------------------
#define NNODES 100000
#define NEDGES 3200000
#define INFEAT 512
#define HID    64
#define NREL   8
#define LATENT 32

typedef __attribute__((ext_vector_type(16))) __bf16 v16bf;
typedef __attribute__((ext_vector_type(8)))  float  v8f;

__device__ __forceinline__ unsigned short f32_to_bf16(float f) {
    unsigned int u = __float_as_uint(f);
    u += 0x7FFFu + ((u >> 16) & 1u);       // round-to-nearest-even
    return (unsigned short)(u >> 16);
}

// ---------------------------------------------------------------------------
// Pre-pass: f32 -> bf16 conversions / weight transposes
// ---------------------------------------------------------------------------
__global__ void k_convert_bf16(const float* __restrict__ in,
                               unsigned short* __restrict__ out, long n) {
    long i = (long)blockIdx.x * blockDim.x + threadIdx.x;
    if (i < n) out[i] = f32_to_bf16(in[i]);
}

// W1 [R][IN][HID] -> W1t bf16 [R][HID][IN]   (B columns contiguous in K)
__global__ void k_transpose_w1(const float* __restrict__ in,
                               unsigned short* __restrict__ out) {
    int i = blockIdx.x * blockDim.x + threadIdx.x;
    if (i >= NREL * INFEAT * HID) return;
    int r = i / (INFEAT * HID);
    int rem = i % (INFEAT * HID);
    int d = rem / HID, k = rem % HID;
    out[(long)r * HID * INFEAT + (long)k * INFEAT + d] = f32_to_bf16(in[i]);
}

// W2 [R][HID][HID] -> W2t bf16 [R][HID(out)][HID(in)]
__global__ void k_transpose_w2(const float* __restrict__ in,
                               unsigned short* __restrict__ out) {
    int i = blockIdx.x * blockDim.x + threadIdx.x;
    if (i >= NREL * HID * HID) return;
    int r = i / (HID * HID);
    int rem = i % (HID * HID);
    int k1 = rem / HID, k2 = rem % HID;
    out[(long)r * HID * HID + (long)k2 * HID + k1] = f32_to_bf16(in[i]);
}

// Wm/Ws [R*HID][LATENT] -> bf16 [R][LATENT][HID]
__global__ void k_transpose_wlat(const float* __restrict__ in,
                                 unsigned short* __restrict__ out) {
    int i = blockIdx.x * blockDim.x + threadIdx.x;
    if (i >= NREL * HID * LATENT) return;
    int r = i / (HID * LATENT);
    int rem = i % (HID * LATENT);
    int k = rem / LATENT, l = rem % LATENT;
    out[(long)r * LATENT * HID + (long)l * HID + k] = f32_to_bf16(in[i]);
}

// ---------------------------------------------------------------------------
// WMMA fragment loaders
//  A (16x32 bf16): lane -> row = lane&15 ; kb = 8*(lane>=16)
//     elems 0..7  = K[k0+kb .. k0+kb+7]
//     elems 8..15 = K[k0+kb+16 .. k0+kb+23]
//  B (32x16 bf16, column-contiguous storage): lane -> col = lane&15 ; kb = 8*(lane>=16)
//     elems 0..7  = K[k0+kb .. k0+kb+7]
//     elems 8..15 = K[k0+kb+16 .. k0+kb+23]
// ---------------------------------------------------------------------------
__device__ __forceinline__ v16bf load_frag_a(const unsigned short* row, int k0, int lane) {
    int kb = (lane >> 4) << 3;                 // 0 or 8
    const unsigned short* p = row + k0 + kb;
    v16bf a;
    ((int4*)&a)[0] = *(const int4*)(p);        // K = kb+0..7
    ((int4*)&a)[1] = *(const int4*)(p + 16);   // K = kb+16..23
    return a;
}

__device__ __forceinline__ v16bf load_frag_b(const unsigned short* col, int k0, int lane) {
    int kb = (lane >> 4) << 3;                 // 0 or 8
    const unsigned short* p = col + k0 + kb;
    v16bf b;
    ((int4*)&b)[0] = *(const int4*)(p);        // K = kb+0..7
    ((int4*)&b)[1] = *(const int4*)(p + 16);   // K = kb+16..23
    return b;
}

// ---------------------------------------------------------------------------
// GEMM (per relation):  C[r] = A[r] (Nrows x K, bf16) @ B[r] (K x NT*16, bf16)
// B stored column-contiguous: [r][NT*16][K].  One wave = 16-row output strip.
// ---------------------------------------------------------------------------
template <int NT>
__global__ __launch_bounds__(256)
void k_gemm_bf16(const unsigned short* __restrict__ A, long aRelStride,
                 const unsigned short* __restrict__ B,
                 float* __restrict__ C, int Nrows, int K) {
    const int wave = threadIdx.x >> 5;
    const int lane = threadIdx.x & 31;
    const int r    = blockIdx.y;
    const int tile = blockIdx.x * (blockDim.x >> 5) + wave;
    const int row0 = tile * 16;
    if (row0 >= Nrows) return;                 // whole-wave exit, EXEC stays all-1

    const unsigned short* Ar = A + (long)r * aRelStride;
    const unsigned short* Br = B + (long)r * (NT * 16) * K;
    const int m = lane & 15;
    const unsigned short* arow = Ar + (long)(row0 + m) * K;

    v8f acc[NT] = {};
    for (int k0 = 0; k0 < K; k0 += 32) {
        v16bf a = load_frag_a(arow, k0, lane);
#pragma unroll
        for (int j = 0; j < NT; ++j) {
            v16bf b = load_frag_b(Br + (long)(j * 16 + m) * K, k0, lane);
            acc[j] = __builtin_amdgcn_wmma_f32_16x16x32_bf16(
                false, a, false, b, (short)0, acc[j], false, false);
        }
    }

    float* Cr = C + (long)r * Nrows * (NT * 16);
    const int moff = (lane >> 4) << 3;         // 0 or 8
#pragma unroll
    for (int j = 0; j < NT; ++j)
#pragma unroll
        for (int i = 0; i < 8; ++i)
            Cr[(long)(row0 + moff + i) * (NT * 16) + j * 16 + m] = acc[j][i];
}

// ---------------------------------------------------------------------------
// Latent GEMM:  C (Nrows x 32) = sum_r  HB[r] (Nrows x 64) @ Wt[r] (64 x 32)
// Wt stored [r][32][64] (column-contiguous in K).
// ---------------------------------------------------------------------------
__global__ __launch_bounds__(256)
void k_gemm_latent(const unsigned short* __restrict__ HB,
                   const unsigned short* __restrict__ Wt,
                   float* __restrict__ C, int Nrows) {
    const int wave = threadIdx.x >> 5;
    const int lane = threadIdx.x & 31;
    const int tile = blockIdx.x * (blockDim.x >> 5) + wave;
    const int row0 = tile * 16;
    if (row0 >= Nrows) return;

    const int m = lane & 15;
    v8f acc[2] = {};
    for (int r = 0; r < NREL; ++r) {
        const unsigned short* arow = HB + (long)r * Nrows * HID + (long)(row0 + m) * HID;
        const unsigned short* Br = Wt + (long)r * LATENT * HID;
        for (int k0 = 0; k0 < HID; k0 += 32) {
            v16bf a = load_frag_a(arow, k0, lane);
#pragma unroll
            for (int j = 0; j < 2; ++j) {
                v16bf b = load_frag_b(Br + (long)(j * 16 + m) * HID, k0, lane);
                acc[j] = __builtin_amdgcn_wmma_f32_16x16x32_bf16(
                    false, a, false, b, (short)0, acc[j], false, false);
            }
        }
    }
    const int moff = (lane >> 4) << 3;
#pragma unroll
    for (int j = 0; j < 2; ++j)
#pragma unroll
        for (int i = 0; i < 8; ++i)
            C[(long)(row0 + moff + i) * LATENT + j * 16 + m] = acc[j][i];
}

// ---------------------------------------------------------------------------
// SpMM scatter-add:  out[r][dst][f] += w[e] * feat[r][src][f]
// F threads per edge; hardware float atomics (L2-resident feature table).
// ---------------------------------------------------------------------------
__global__ __launch_bounds__(256)
void k_spmm(const int* __restrict__ src, const int* __restrict__ dst,
            const float* __restrict__ w,
            const float* __restrict__ feat, float* __restrict__ out,
            int F, long relStride, int E) {
    const int r = blockIdx.y;
    const int perBlock = blockDim.x / F;
    const int e = blockIdx.x * perBlock + (int)(threadIdx.x / F);
    const int f = threadIdx.x % F;
    if (e >= E) return;
    const int s = src[e], d = dst[e];
    const float v = w[e] * feat[(long)r * relStride + (long)s * F + f];
    atomicAdd(&out[(long)r * relStride + (long)d * F + f], v);
}

// tanh(agg) -> bf16 for the next WMMA stage
__global__ void k_tanh_bf16(const float* __restrict__ in,
                            unsigned short* __restrict__ out, long n) {
    long i = (long)blockIdx.x * blockDim.x + threadIdx.x;
    if (i < n) out[i] = f32_to_bf16(tanhf(in[i]));
}

// z = eps*std + m ; outputs concatenated (z, m, std)
__global__ void k_final(const float* __restrict__ aggm, const float* __restrict__ aggs,
                        const float* __restrict__ eps, float* __restrict__ out, long nl) {
    long i = (long)blockIdx.x * blockDim.x + threadIdx.x;
    if (i >= nl) return;
    float m = aggm[i];
    float s = fmaxf(aggs[i], 0.0f) + 1.0e-4f;
    out[i]          = eps[i] * s + m;
    out[nl + i]     = m;
    out[2 * nl + i] = s;
}

// ---------------------------------------------------------------------------
// Host orchestration
// ---------------------------------------------------------------------------
extern "C" void kernel_launch(void* const* d_in, const int* in_sizes, int n_in,
                              void* d_out, int out_size, void* d_ws, size_t ws_size,
                              hipStream_t stream) {
    const float* x   = (const float*)d_in[0];
    const int*   esr = (const int*)d_in[1];
    const int*   eds = (const int*)d_in[2];
    const float* ew  = (const float*)d_in[3];
    const float* W1  = (const float*)d_in[4];
    const float* W2  = (const float*)d_in[5];
    const float* Wm  = (const float*)d_in[6];
    const float* Ws  = (const float*)d_in[7];
    const float* eps = (const float*)d_in[8];
    float* out = (float*)d_out;

    // ---- workspace carve-out (256B aligned regions) ----
    char* ws = (char*)d_ws;
    size_t off = 0;
    auto carve = [&](size_t bytes) -> void* {
        void* p = ws + off;
        off = (off + bytes + 255) & ~(size_t)255;
        return p;
    };
    unsigned short* XB   = (unsigned short*)carve((size_t)NNODES * INFEAT * 2);      // 102.4 MB
    float*          HPRE = (float*)carve((size_t)NREL * NNODES * HID * 4);           // 204.8 MB
    float*          AGG  = (float*)carve((size_t)NREL * NNODES * HID * 4);           // 204.8 MB
    unsigned short* HB   = (unsigned short*)carve((size_t)NREL * NNODES * HID * 2);  // 102.4 MB
    unsigned short* W1T  = (unsigned short*)carve((size_t)NREL * HID * INFEAT * 2);
    unsigned short* W2T  = (unsigned short*)carve((size_t)NREL * HID * HID * 2);
    unsigned short* WMT  = (unsigned short*)carve((size_t)NREL * LATENT * HID * 2);
    unsigned short* WST  = (unsigned short*)carve((size_t)NREL * LATENT * HID * 2);
    float*          ZPM  = (float*)carve((size_t)NNODES * LATENT * 4);
    float*          ZPS  = (float*)carve((size_t)NNODES * LATENT * 4);
    float*          ZAM  = (float*)carve((size_t)NNODES * LATENT * 4);
    float*          ZAS  = (float*)carve((size_t)NNODES * LATENT * 4);
    (void)ws_size; (void)n_in; (void)in_sizes; (void)out_size;

    const int TB = 256;
    const long nX = (long)NNODES * INFEAT;
    const long nH = (long)NREL * NNODES * HID;
    const long nL = (long)NNODES * LATENT;
    const int rowTiles = NNODES / 16;                         // 6250 (exact)
    const dim3 gGemm((rowTiles + 7) / 8, NREL);
    const dim3 gGemmL((rowTiles + 7) / 8, 1);
    const dim3 gSpmm64((NEDGES + 3) / 4, NREL);               // 4 edges / block @ F=64
    const dim3 gSpmm32((NEDGES + 7) / 8, 1);                  // 8 edges / block @ F=32

    // -- pre-pass: precision conversion & weight transposes
    k_convert_bf16<<<(unsigned)((nX + TB - 1) / TB), TB, 0, stream>>>(x, XB, nX);
    k_transpose_w1<<<(NREL * INFEAT * HID + TB - 1) / TB, TB, 0, stream>>>(W1, W1T);
    k_transpose_w2<<<(NREL * HID * HID + TB - 1) / TB, TB, 0, stream>>>(W2, W2T);
    k_transpose_wlat<<<(NREL * HID * LATENT + TB - 1) / TB, TB, 0, stream>>>(Wm, WMT);
    k_transpose_wlat<<<(NREL * HID * LATENT + TB - 1) / TB, TB, 0, stream>>>(Ws, WST);

    // -- layer 1: h1 = tanh(adj @ (x @ W1_r))
    k_gemm_bf16<4><<<gGemm, TB, 0, stream>>>(XB, 0L, W1T, HPRE, NNODES, INFEAT);
    hipMemsetAsync(AGG, 0, (size_t)nH * 4, stream);
    k_spmm<<<gSpmm64, TB, 0, stream>>>(esr, eds, ew, HPRE, AGG, HID, (long)NNODES * HID, NEDGES);
    k_tanh_bf16<<<(unsigned)((nH + TB - 1) / TB), TB, 0, stream>>>(AGG, HB, nH);

    // -- layer 2: h2 = tanh(adj @ (h1 @ W2_r))
    k_gemm_bf16<4><<<gGemm, TB, 0, stream>>>(HB, (long)NNODES * HID, W2T, HPRE, NNODES, HID);
    hipMemsetAsync(AGG, 0, (size_t)nH * 4, stream);
    k_spmm<<<gSpmm64, TB, 0, stream>>>(esr, eds, ew, HPRE, AGG, HID, (long)NNODES * HID, NEDGES);
    k_tanh_bf16<<<(unsigned)((nH + TB - 1) / TB), TB, 0, stream>>>(AGG, HB, nH);

    // -- latent heads: Z@Wm, Z@Ws  (= sum over relations, fused in WMMA accumulators)
    k_gemm_latent<<<gGemmL, TB, 0, stream>>>(HB, WMT, ZPM, NNODES);
    k_gemm_latent<<<gGemmL, TB, 0, stream>>>(HB, WST, ZPS, NNODES);

    // -- final aggregation + reparameterize
    hipMemsetAsync(ZAM, 0, (size_t)nL * 4, stream);
    hipMemsetAsync(ZAS, 0, (size_t)nL * 4, stream);
    k_spmm<<<gSpmm32, TB, 0, stream>>>(esr, eds, ew, ZPM, ZAM, LATENT, 0L, NEDGES);
    k_spmm<<<gSpmm32, TB, 0, stream>>>(esr, eds, ew, ZPS, ZAS, LATENT, 0L, NEDGES);
    k_final<<<(unsigned)((nL + TB - 1) / TB), TB, 0, stream>>>(ZAM, ZAS, eps, out, nL);
}
